// GraphSAGE_29712583754274
// MI455X (gfx1250) — matrix operand order repaired and checked by reading, not compiled, not run on hardware
//
#include <hip/hip_runtime.h>

#define NN   50000
#define NE   500000
#define DIM  512
#define NNH  (NN * DIM)   // 25,600,000 floats per activation matrix

typedef __attribute__((ext_vector_type(16))) _Float16 v16h;
typedef __attribute__((ext_vector_type(8)))  float    v8f;

union FragAB { v16h h; unsigned u[8]; };
union FragC  { v8f  v; float    f[8]; };

__device__ inline unsigned pack2(float a, float b) {
    union { _Float16 h[2]; unsigned u; } t;
    t.h[0] = (_Float16)a;
    t.h[1] = (_Float16)b;
    return t.u;
}

// ---------------------------------------------------------------------------
// Edge scatter: 128 threads per edge, each handles 4 contiguous floats.
// sum[] (102 MB) is L2-resident, so the f32 atomics stay on-chip.
// ---------------------------------------------------------------------------
__global__ __launch_bounds__(256) void scatter_add(
    const float* __restrict__ feat, const int* __restrict__ src,
    const int* __restrict__ dst, float* __restrict__ sum,
    float* __restrict__ deg, int nE)
{
    int tid = blockIdx.x * blockDim.x + threadIdx.x;
    int e = tid >> 7;
    if (e >= nE) return;
    int j = tid & 127;
    int s = src[e], d = dst[e];
    const float4 v = *((const float4*)(feat + (size_t)s * DIM) + j);
    float* p = sum + (size_t)d * DIM + j * 4;
    atomicAdd(p + 0, v.x);
    atomicAdd(p + 1, v.y);
    atomicAdd(p + 2, v.z);
    atomicAdd(p + 3, v.w);
    if (j == 0) atomicAdd(deg + d, 1.0f);
}

// mean = sum / max(deg, 1)
__global__ __launch_bounds__(256) void mean_div(
    float* __restrict__ sum, const float* __restrict__ deg, int n)
{
    int tid = blockIdx.x * blockDim.x + threadIdx.x;
    int node = tid >> 7;
    if (node >= n) return;
    float d = deg[node];
    d = (d < 1.0f) ? 1.0f : d;
    float inv = 1.0f / d;
    float4* p = (float4*)(sum + (size_t)node * DIM) + (tid & 127);
    float4 v = *p;
    v.x *= inv; v.y *= inv; v.z *= inv; v.w *= inv;
    *p = v;
}

// Transpose + fp32->f16 convert one 512x512 weight: Wt[n][k] = W[k][n].
__global__ __launch_bounds__(256) void convert_w(
    const float* __restrict__ W, _Float16* __restrict__ Wt)
{
    int tid = blockIdx.x * 256 + threadIdx.x;   // 262144 threads
    int n = tid >> 9;
    int k = tid & 511;
    Wt[(size_t)n * DIM + k] = (_Float16)W[(size_t)k * DIM + n];
}

// ---------------------------------------------------------------------------
// Fused dual GEMM: out = act(A1@W1 + A2@W2 + bias), K = 512 per part, N = 512.
// 256 threads = 8 waves; block tile 128x128; wave tile 32x64 (2x4 WMMAs).
// A: fp32 -> f16 packed into double-buffered LDS (17-uint row stride ->
//    17*m mod 32 covers all banks, conflict-free fragment gathers).
// B: pre-converted f16 W^T in global (L2-resident); each lane's fragment is
//    two 4-uint runs at +0 and +8 uints -> 2x global_load_b128, no LDS, no barrier coupling.
// One s_barrier per k-step via LDS double buffering.
// ---------------------------------------------------------------------------
__global__ __launch_bounds__(256) void sage_gemm(
    const float* __restrict__ A1, const unsigned* __restrict__ Wt1,
    const float* __restrict__ A2, const unsigned* __restrict__ Wt2,
    const float* __restrict__ bias, float* __restrict__ out,
    int M, int kParts, int doRelu, int ntStore)
{
    __shared__ unsigned ldsA[2][128 * 17];  // 2 x (128 rows x 32 halves + pad)

    const int tid   = threadIdx.x;
    const int lane  = tid & 31;
    const int wave  = tid >> 5;
    const int waveM = wave & 3;       // 4 waves along M (32 rows each)
    const int waveN = wave >> 2;      // 2 waves along N (64 cols each)
    const int colBase = blockIdx.x * 128;   // x fastest => row-chunk L2 reuse
    const int rowBase = blockIdx.y * 128;

    FragC acc[2][4];
#pragma unroll
    for (int i = 0; i < 2; ++i)
#pragma unroll
        for (int j = 0; j < 4; ++j)
#pragma unroll
            for (int r = 0; r < 8; ++r) acc[i][j].f[r] = 0.0f;

    // A staging: thread -> (row sr, 16-half chunk sp). Rows >= M are clamped;
    // their garbage only reaches output rows >= M, which are never stored.
    const int sr = tid >> 1;
    const int sp = tid & 1;
    const int grow = min(rowBase + sr, M - 1);
    const float* rowPtr1 = A1 + (size_t)grow * DIM + sp * 16;
    const float* rowPtr2 = A2 + (size_t)grow * DIM + sp * 16;

    auto stageA = [&](int buf, int kt) {
        const float* base = (kt < 16) ? rowPtr1 : rowPtr2;
        const float4* g = (const float4*)(base + (kt & 15) * 32);
        float4 f0 = g[0], f1 = g[1], f2 = g[2], f3 = g[3];
        unsigned* d = &ldsA[buf][sr * 17 + sp * 8];
        d[0] = pack2(f0.x, f0.y); d[1] = pack2(f0.z, f0.w);
        d[2] = pack2(f1.x, f1.y); d[3] = pack2(f1.z, f1.w);
        d[4] = pack2(f2.x, f2.y); d[5] = pack2(f2.z, f2.w);
        d[6] = pack2(f3.x, f3.y); d[7] = pack2(f3.z, f3.w);
    };

    const int ktTotal = kParts * 16;
    const int m16 = lane & 15;
    const int kg  = lane >> 4;

    int cur = 0;
    stageA(0, 0);

    for (int kt = 0; kt < ktTotal; ++kt) {
        __syncthreads();
        if (kt + 1 < ktTotal) stageA(cur ^ 1, kt + 1);

        const int k0 = (kt & 15) * 32;
        const unsigned* Wt = (kt < 16) ? Wt1 : Wt2;

        FragAB a[2], b[4];
#pragma unroll
        for (int mi = 0; mi < 2; ++mi) {
            int row = waveM * 32 + mi * 16 + m16;
#pragma unroll
            for (int v = 0; v < 8; ++v) {
                // A 16-bit 16x32 layout: VGPR v holds a K pair; lanes>=16 +8
                int colu = (v & 3) + ((v >> 2) * 8) + kg * 4;
                a[mi].u[v] = ldsA[cur][row * 17 + colu];
            }
        }
#pragma unroll
        for (int ni = 0; ni < 4; ++ni) {
            int n = colBase + waveN * 64 + ni * 16 + m16;
            const uint4* p =
                (const uint4*)(Wt + (size_t)n * 256 + (k0 >> 1) + kg * 4);
            uint4 q0 = p[0], q1 = p[2];
            b[ni].u[0] = q0.x; b[ni].u[1] = q0.y;
            b[ni].u[2] = q0.z; b[ni].u[3] = q0.w;
            b[ni].u[4] = q1.x; b[ni].u[5] = q1.y;
            b[ni].u[6] = q1.z; b[ni].u[7] = q1.w;
        }
#pragma unroll
        for (int mi = 0; mi < 2; ++mi)
#pragma unroll
            for (int ni = 0; ni < 4; ++ni)
                acc[mi][ni].v = __builtin_amdgcn_wmma_f32_16x16x32_f16(
                    false, a[mi].h, false, b[ni].h,
                    (short)0, acc[mi][ni].v, false, false);
        cur ^= 1;
    }

    // ---- epilogue: bias + optional relu, fp32 store ----
    const int nIdx = lane & 15;
    const int lg   = lane >> 4;
#pragma unroll
    for (int mi = 0; mi < 2; ++mi) {
#pragma unroll
        for (int ni = 0; ni < 4; ++ni) {
            int col = colBase + waveN * 64 + ni * 16 + nIdx;
            float bv = bias[col];
#pragma unroll
            for (int r = 0; r < 8; ++r) {
                int row = rowBase + waveM * 32 + mi * 16 + r + lg * 8;
                if (row < M) {
                    float vv = acc[mi][ni].f[r] + bv;
                    if (doRelu) vv = vv > 0.0f ? vv : 0.0f;
                    float* dst = &out[(size_t)row * DIM + col];
                    if (ntStore) __builtin_nontemporal_store(vv, dst);
                    else *dst = vv;
                }
            }
        }
    }
}

// ---------------------------------------------------------------------------
extern "C" void kernel_launch(void* const* d_in, const int* in_sizes, int n_in,
                              void* d_out, int out_size, void* d_ws, size_t ws_size,
                              hipStream_t stream)
{
    (void)in_sizes; (void)n_in; (void)out_size; (void)ws_size;

    const float* x   = (const float*)d_in[0];
    const int*   ei  = (const int*)d_in[1];
    const float* Wl0 = (const float*)d_in[2];
    const float* bl0 = (const float*)d_in[3];
    const float* Wr0 = (const float*)d_in[4];
    const float* Wl1 = (const float*)d_in[5];
    const float* bl1 = (const float*)d_in[6];
    const float* Wr1 = (const float*)d_in[7];
    const float* Wv  = (const float*)d_in[8];
    const float* bv  = (const float*)d_in[9];
    const float* Wt  = (const float*)d_in[10];
    const float* bt  = (const float*)d_in[11];

    const int* src = ei;          // edge_index[0]
    const int* dst = ei + NE;     // edge_index[1]

    float* mean = (float*)d_ws;                    // NNH floats
    float* deg  = mean + (size_t)NNH;              // NN floats (+pad)
    float* h1   = deg + 51200;                     // NNH floats
    _Float16* w16base = (_Float16*)(h1 + (size_t)NNH);  // 6 x 512x512 halves
    const size_t WSZ = (size_t)DIM * DIM;
    _Float16* wl0t = w16base + 0 * WSZ;
    _Float16* wr0t = w16base + 1 * WSZ;
    _Float16* wl1t = w16base + 2 * WSZ;
    _Float16* wr1t = w16base + 3 * WSZ;
    _Float16* wvt  = w16base + 4 * WSZ;
    _Float16* wtt  = w16base + 5 * WSZ;

    float* h2 = (float*)d_out;                  // output 0
    float* xv = h2 + (size_t)NNH;               // output 1
    float* xt = xv + (size_t)NNH;               // output 2

    dim3 gemmGrid(DIM / 128, (NN + 127) / 128); // col tile fastest -> L2 reuse
    dim3 gemmBlock(256);
    const int scatterBlocks = (NE * 128) / 256; // 250000
    const int divBlocks     = (NN * 128) / 256; // 25000
    const int cvtBlocks     = (int)(WSZ / 256); // 1024

    // ---- pre-convert all weights to f16 W^T (3 MB total, L2-resident) ----
    convert_w<<<cvtBlocks, 256, 0, stream>>>(Wl0, wl0t);
    convert_w<<<cvtBlocks, 256, 0, stream>>>(Wr0, wr0t);
    convert_w<<<cvtBlocks, 256, 0, stream>>>(Wl1, wl1t);
    convert_w<<<cvtBlocks, 256, 0, stream>>>(Wr1, wr1t);
    convert_w<<<cvtBlocks, 256, 0, stream>>>(Wv,  wvt);
    convert_w<<<cvtBlocks, 256, 0, stream>>>(Wt,  wtt);

    // ---- layer 0: h1 = relu(mean(x) @ Wl0 + bl0 + x @ Wr0) ----
    hipMemsetAsync(mean, 0, (size_t)(NNH + NN) * sizeof(float), stream);
    scatter_add<<<scatterBlocks, 256, 0, stream>>>(x, src, dst, mean, deg, NE);
    mean_div<<<divBlocks, 256, 0, stream>>>(mean, deg, NN);
    sage_gemm<<<gemmGrid, gemmBlock, 0, stream>>>(
        mean, (const unsigned*)wl0t, x, (const unsigned*)wr0t,
        bl0, h1, NN, 2, 1, 0);

    // ---- layer 1: h2 = mean(h1) @ Wl1 + bl1 + h1 @ Wr1 (no relu) ----
    hipMemsetAsync(mean, 0, (size_t)(NNH + NN) * sizeof(float), stream);
    scatter_add<<<scatterBlocks, 256, 0, stream>>>(h1, src, dst, mean, deg, NE);
    mean_div<<<divBlocks, 256, 0, stream>>>(mean, deg, NN);
    sage_gemm<<<gemmGrid, gemmBlock, 0, stream>>>(
        mean, (const unsigned*)wl1t, h1, (const unsigned*)wr1t,
        bl1, h2, NN, 2, 0, 0);

    // ---- heads: x_vision = relu(h2@Wv+bv), x_text = relu(h2@Wt+bt) ----
    sage_gemm<<<gemmGrid, gemmBlock, 0, stream>>>(
        h2, (const unsigned*)wvt, h2, (const unsigned*)wvt,
        bv, xv, NN, 1, 1, 1);
    sage_gemm<<<gemmGrid, gemmBlock, 0, stream>>>(
        h2, (const unsigned*)wtt, h2, (const unsigned*)wtt,
        bt, xt, NN, 1, 1, 1);
}
